// ColorDenseCRFLoss_40175124086814
// MI455X (gfx1250) — compile-verified, matrix operand order checked by reading, not run on hardware
//
#include <hip/hip_runtime.h>
#include <hip/hip_bf16.h>

typedef __attribute__((ext_vector_type(16))) _Float16 v16h;
typedef __attribute__((ext_vector_type(8)))  _Float16 v8h;
typedef __attribute__((ext_vector_type(8)))  float    v8f;
typedef __attribute__((ext_vector_type(2)))  float    v2f;

#define IMH 128
#define IMW 128
#define OH   64
#define OW   64
#define PPIX (OH*OW)        // 4096 pixels per image
#define TILES (PPIX/16)     // 256 tiles of 16 pixels
#define KMAX 32             // classes padded 21 -> 32 for WMMA K dim
#define SIGMA_RGB 15.0f
#define LOG2E 1.44269504088896340736f

// ---------------- Kernel 1: downsample + pack (exp2-domain prescale) --------
// feats4[(n*P+p)*4] : (r,g,b,0)/15 * sqrt(log2 e)   -> WMMA dot = log2(e)*f.f
// sqv[n*P+p]        : 0.5*|f'|^2                    -> exponent adds, no muls
// segT[(n*P+p)*32]  : 2x2-avg-pooled softmax probs as f16, classes 21..31 = 0
__global__ void prep_kernel(const float* __restrict__ images,
                            const float* __restrict__ segs,
                            int N, int KC,
                            float4* __restrict__ feats4,
                            float*  __restrict__ sqv,
                            _Float16* __restrict__ segT) {
  int gid = blockIdx.x * blockDim.x + threadIdx.x;
  if (gid >= N * PPIX) return;
  int n = gid / PPIX, p = gid % PPIX;
  int y = p / OW, x = p % OW;
  int yy = 2 * y, xx = 2 * x;

  const float scale = 0.0800759014f; // sqrt(log2 e) / 15
  const float* ib = images + (((size_t)n * 3) * IMH + yy) * IMW + xx;
  float4 f;
  f.x = ib[0]                      * scale;
  f.y = ib[(size_t)IMH * IMW]      * scale;
  f.z = ib[2 * (size_t)IMH * IMW]  * scale;
  f.w = 0.0f;
  feats4[gid] = f;
  sqv[gid] = 0.5f * (f.x * f.x + f.y * f.y + f.z * f.z);

  _Float16* st = segT + (size_t)gid * KMAX;
  for (int k = 0; k < KMAX; ++k) {
    float s = 0.0f;
    if (k < KC) {
      const float* sb = segs + ((((size_t)n * KC + k) * IMH + yy) * IMW + xx);
      s = 0.25f * (sb[0] + sb[1] + sb[IMW] + sb[IMW + 1]);  // exact bilinear 2x down
    }
    st[k] = (_Float16)s;
  }
}

// ---------------- Kernel 2: tiled bilateral quadratic form (upper triangle) --
// Per wave-uniform row tile I, accumulate sum over J >= I tiles of
//   wgt * < G_IJ , K_IJ >,  wgt = (J==I ? 1 : 2)   [K and G are symmetric]
__device__ __forceinline__ void process_row(
    int I, int wave, int half, int col, size_t base,
    const float* __restrict__ feats,   // float view of feats4
    const float* __restrict__ sqv,
    const _Float16* __restrict__ segT,
    float& acc) {
  const int pi = I * 16 + col;

  // A operands: address-folded 'half' selection (no cndmask, minimal bytes)
  const float2 fa = *(const float2*)(feats + (size_t)(base + pi) * 4 + 2 * half);
  v2f Af; Af.x = fa.x; Af.y = fa.y;                     // K pair {0,1} or {2,3}
  const _Float16* sa = segT + (base + pi) * KMAX + 8 * half;
  v16h Aseg = __builtin_shufflevector(*(const v8h*)sa, *(const v8h*)(sa + 16),
                                      0,1,2,3,4,5,6,7,8,9,10,11,12,13,14,15);
  float sqI[8];
  #pragma unroll
  for (int r = 0; r < 8; ++r)                           // C/D row r -> r + 8*half
    sqI[r] = sqv[base + I * 16 + r + 8 * half];         // already 0.5*log2e*|f|^2

  const v8f zero = {};
  for (int J = I + wave; J < TILES; J += 8) {           // uniform per wave
    const float wgt = (J == I) ? 1.0f : 2.0f;           // wave-uniform
    const int pj = J * 16 + col;

    const float2 fb = *(const float2*)(feats + (size_t)(base + pj) * 4 + 2 * half);
    v2f Bf; Bf.x = fb.x; Bf.y = fb.y;
    const _Float16* sb = segT + (base + pj) * KMAX + 8 * half;
    v16h Bseg = __builtin_shufflevector(*(const v8h*)sb, *(const v8h*)(sb + 16),
                                        0,1,2,3,4,5,6,7,8,9,10,11,12,13,14,15);
    float sqJ = sqv[base + pj];                         // column term: col

    if (J + 8 < TILES)                                  // L2 latency hiding
      __builtin_prefetch(segT + (base + (J + 8) * 16 + col) * KMAX, 0, 0);

    // dot2[i][j] = log2(e) * f_i.f_j   (16x16x4 f32, prescaled features)
    v8f dotm = __builtin_amdgcn_wmma_f32_16x16x4_f32(
        false, Af, false, Bf, (short)0, zero, false, false);
    // G[i][j] = sum_k S[k][pi] * S[k][pj]   (16x16x32 f16 -> f32)
    v8f g = __builtin_amdgcn_wmma_f32_16x16x32_f16(
        false, Aseg, false, Bseg, (short)0, zero, false, false);

    float t = 0.0f;
    #pragma unroll
    for (int r = 0; r < 8; ++r) {
      // exp(-0.5*max(d2,0)) == exp2(min(dot2 - sqI - sqJ, 0)) in exp2 domain
      float e = fminf(dotm[r] - sqI[r] - sqJ, 0.0f);
      t += g[r] * __builtin_amdgcn_exp2f(e);            // native v_exp_f32
    }
    acc += wgt * t;
  }
}

__global__ void __launch_bounds__(256)
crf_tile_kernel(const float* __restrict__ feats,
                const float* __restrict__ sqv,
                const _Float16* __restrict__ segT,
                float* __restrict__ partials) {
  const int wg   = blockIdx.x;
  const int n    = wg / (TILES / 2);
  const int g    = wg % (TILES / 2);     // row pair (g, 255-g): 257 tiles, balanced
  const int tid  = threadIdx.x;
  const int lane = tid & 31;
  const int wave = tid >> 5;
  const int half = lane >> 4;
  const int col  = lane & 15;
  const size_t base = (size_t)n * PPIX;

  float acc = 0.0f;
  process_row(g,             wave, half, col, base, feats, sqv, segT, acc);
  process_row(TILES - 1 - g, wave, half, col, base, feats, sqv, segT, acc);

  // wave32 reduce, then cross-wave via LDS (deterministic; no atomics)
  #pragma unroll
  for (int m = 16; m; m >>= 1) acc += __shfl_xor(acc, m, 32);
  __shared__ float wsum[8];
  if (lane == 0) wsum[wave] = acc;
  __syncthreads();
  if (tid == 0) {
    float t = 0.0f;
    for (int i = 0; i < 8; ++i) t += wsum[i];
    partials[wg] = t;
  }
}

// ---------------- Kernel 3: finalize ----------------
__global__ void finalize_kernel(const float* __restrict__ partials, int npart,
                                int N, float* __restrict__ out) {
  float a = 0.0f;
  for (int i = threadIdx.x; i < npart; i += blockDim.x) a += partials[i];
  #pragma unroll
  for (int m = 16; m; m >>= 1) a += __shfl_xor(a, m, 32);
  __shared__ float w[8];
  const int lane = threadIdx.x & 31, wave = threadIdx.x >> 5;
  if (lane == 0) w[wave] = a;
  __syncthreads();
  if (threadIdx.x == 0) {
    float t = 0.0f;
    for (int i = 0; i < (int)(blockDim.x >> 5); ++i) t += w[i];
    out[0] = -(1e-7f) * t / (float)N;   // WEIGHT * (-sum / N)
  }
}

extern "C" void kernel_launch(void* const* d_in, const int* in_sizes, int n_in,
                              void* d_out, int out_size, void* d_ws, size_t ws_size,
                              hipStream_t stream) {
  const float* images = (const float*)d_in[0];
  const float* segs   = (const float*)d_in[1];
  const int N  = in_sizes[0] / (3 * IMH * IMW);          // 4
  const int KC = in_sizes[1] / (N * IMH * IMW);          // 21

  char* ws = (char*)d_ws;
  size_t o = 0;
  float4*   feats4   = (float4*)(ws + o);  o += (size_t)N * PPIX * sizeof(float4);    // 256 KB
  float*    sqv      = (float*)(ws + o);   o += (size_t)N * PPIX * sizeof(float);     //  64 KB
  _Float16* segT     = (_Float16*)(ws + o);o += (size_t)N * PPIX * KMAX * sizeof(_Float16); // 1 MB
  float*    partials = (float*)(ws + o);                                              //   2 KB

  const int total = N * PPIX;
  prep_kernel<<<(total + 255) / 256, 256, 0, stream>>>(images, segs, N, KC,
                                                       feats4, sqv, segT);
  const int nblocks = N * (TILES / 2);                   // 512 balanced blocks
  crf_tile_kernel<<<nblocks, 256, 0, stream>>>((const float*)feats4, sqv, segT,
                                               partials);
  finalize_kernel<<<1, 256, 0, stream>>>(partials, nblocks, N, (float*)d_out);
}
